// ResGCN_89240830476623
// MI455X (gfx1250) — matrix operand order repaired and checked by reading, not compiled, not run on hardware
//
#include <hip/hip_runtime.h>

typedef __attribute__((ext_vector_type(16))) _Float16 v16h;
typedef __attribute__((ext_vector_type(8)))  _Float16 v8h;
typedef __attribute__((ext_vector_type(8)))  float    v8f;

#define HD 128

// ---------------------------------------------------------------- utilities
__global__ void k_zero(float* __restrict__ p, long long n) {
  long long i = (long long)blockIdx.x * blockDim.x + threadIdx.x;
  if (i < n) p[i] = 0.f;
}

__global__ void k_edge_deg(const int* __restrict__ dst, float* __restrict__ deg, int E) {
  int e = blockIdx.x * blockDim.x + threadIdx.x;
  if (e < E) atomicAdd(&deg[dst[e]], 1.0f);
}

__global__ void k_dinv(float* __restrict__ d, int N) {
  int i = blockIdx.x * blockDim.x + threadIdx.x;
  if (i < N) d[i] = rsqrtf(d[i] + 1.0f);
}

// ------------------------------------------------------------ BatchNorm (N)
__global__ void k_bn_stats(const float* __restrict__ h, float* __restrict__ sums,
                           int N, int rowsPerBlock) {
  int c = threadIdx.x;                       // 0..127
  int r0 = blockIdx.x * rowsPerBlock;
  int r1 = r0 + rowsPerBlock; if (r1 > N) r1 = N;
  float s = 0.f, s2 = 0.f;
  for (int r = r0; r < r1; ++r) {
    float v = h[(long long)r * HD + c];
    s += v; s2 += v * v;
  }
  atomicAdd(&sums[c], s);
  atomicAdd(&sums[HD + c], s2);
}

__global__ void k_bn_finalize(float* __restrict__ ss, const float* __restrict__ g,
                              const float* __restrict__ b, float invN) {
  int c = threadIdx.x;
  float mean = ss[c] * invN;
  float var  = ss[HD + c] * invN - mean * mean;   // biased batch variance
  float sc = g[c] * rsqrtf(var + 1e-5f);
  ss[c]      = sc;
  ss[HD + c] = b[c] - mean * sc;
}

__global__ void k_bn_apply_f16(const float* __restrict__ src, const float* __restrict__ ss,
                               _Float16* __restrict__ dst, int N, int Npad) {
  long long i = (long long)blockIdx.x * blockDim.x + threadIdx.x;
  if (i >= (long long)Npad * HD) return;
  int c = (int)(i & (HD - 1));
  int n = (int)(i >> 7);
  float v = (n < N) ? (src[i] * ss[c] + ss[HD + c]) : 0.f;
  dst[i] = (_Float16)v;
}

__global__ void k_to_f16(const float* __restrict__ src, _Float16* __restrict__ dst,
                         int N, int Npad) {
  long long i = (long long)blockIdx.x * blockDim.x + threadIdx.x;
  if (i >= (long long)Npad * HD) return;
  int n = (int)(i >> 7);
  dst[i] = (_Float16)((n < N) ? src[i] : 0.f);
}

// transpose-convert a 128x128 f32 weight to f16, Wt[n][k] = W[k][n]
__global__ void k_wcvt(const float* __restrict__ W, _Float16* __restrict__ Wt) {
  int i = blockIdx.x * blockDim.x + threadIdx.x;   // 16384
  int n = i & (HD - 1), k = i >> 7;
  Wt[n * HD + k] = (_Float16)W[k * HD + n];
}

// ------------------------------------------------------- WMMA GEMM (x @ W)
// A: f16 [rows x 128] (rows multiple of 128), Wt: f16 transposed [128 x 128]
// out: f32 [rows x 128] ; optional bias[128] and ReLU fused.
__global__ __launch_bounds__(256) void k_gemm_wmma(
    const _Float16* __restrict__ A, const _Float16* __restrict__ Wt,
    const float* __restrict__ bias, float* __restrict__ out, int relu) {
  __shared__ __align__(16) _Float16 sW[HD * HD];
  {
    const uint4* src = (const uint4*)Wt;
    uint4* dst = (uint4*)sW;
    #pragma unroll
    for (int i = threadIdx.x; i < HD * HD / 8; i += 256) dst[i] = src[i];
  }
  __syncthreads();

  int wid  = threadIdx.x >> 5;
  int lane = threadIdx.x & 31;
  int m    = lane & 15;        // row within 16-row tile (A), col within tile (B/D)
  int half = lane >> 4;        // lane half selects K sub-ranges
  long long rowBase = (long long)blockIdx.x * 128 + wid * 16;
  const _Float16* arow = A + (rowBase + m) * HD;

  v8f zero = {};
  v8f acc[8];
  #pragma unroll
  for (int nt = 0; nt < 8; ++nt) acc[nt] = zero;

  #pragma unroll
  for (int kt = 0; kt < 4; ++kt) {
    // A 16x32 f16 layout: lanes 0-15 hold K 0-7 (v0..3) and 16-23 (v4..7);
    // lanes 16-31 hold K 8-15 and 24-31.
    int ka = kt * 32 + half * 8;
    v8h a0 = *(const v8h*)(arow + ka);
    v8h a1 = *(const v8h*)(arow + ka + 16);
    v16h a = __builtin_shufflevector(a0, a1,
             0,1,2,3,4,5,6,7,8,9,10,11,12,13,14,15);
    // B 32x16 f16 layout: lanes 0-15 col n K 0-7 and 16-23, lanes 16-31 col n K 8-15 and 24-31.
    int kb = kt * 32 + half * 8;
    #pragma unroll
    for (int nt = 0; nt < 8; ++nt) {
      const _Float16* bp = sW + (nt * 16 + m) * HD + kb;
      v8h b0 = *(const v8h*)bp;
      v8h b1 = *(const v8h*)(bp + 16);
      v16h b = __builtin_shufflevector(b0, b1,
               0,1,2,3,4,5,6,7,8,9,10,11,12,13,14,15);
      acc[nt] = __builtin_amdgcn_wmma_f32_16x16x32_f16(
          false, a, false, b, (short)0, acc[nt], false, false);
    }
  }

  // D f32 layout: VGPR r -> row r (lanes 0-15) / r+8 (lanes 16-31), col = lane&15
  #pragma unroll
  for (int nt = 0; nt < 8; ++nt) {
    int col = nt * 16 + m;
    float bb = bias ? bias[col] : 0.f;
    #pragma unroll
    for (int r = 0; r < 8; ++r) {
      float v = acc[nt][r] + bb;
      if (relu) v = fmaxf(v, 0.f);
      out[(rowBase + half * 8 + r) * HD + col] = v;
    }
  }
}

// ---------------------------------------------------- edge scatter (A-hat)
__global__ void k_scatter(const int* __restrict__ src, const int* __restrict__ dst,
                          const float* __restrict__ dinv, const float* __restrict__ t,
                          float* __restrict__ agg, int E) {
  long long idx = (long long)blockIdx.x * blockDim.x + threadIdx.x;
  int e = (int)(idx >> 5);
  if (e >= E) return;
  int f = ((int)idx & 31) << 2;
  int s = src[e], d = dst[e];
  float coef = dinv[s] * dinv[d];
  const float4 v = *(const float4*)(t + (long long)s * HD + f);
  float* o = agg + (long long)d * HD + f;
  atomicAdd(o + 0, v.x * coef);
  atomicAdd(o + 1, v.y * coef);
  atomicAdd(o + 2, v.z * coef);
  atomicAdd(o + 3, v.w * coef);
}

__global__ void k_finalize(float* __restrict__ h, const float* __restrict__ t,
                           const float* __restrict__ dinv, const float* __restrict__ bias,
                           int N) {
  long long i = (long long)blockIdx.x * blockDim.x + threadIdx.x;
  if (i >= (long long)N * HD) return;
  int n = (int)(i >> 7), c = (int)(i & (HD - 1));
  float di = dinv[n];
  float v = h[i] + t[i] * di * di + bias[c];
  h[i] = fmaxf(v, 0.f);
}

// ------------------------------------------------------------ gate + pool
__global__ void k_gate_dot(const float* __restrict__ u, const float* __restrict__ w2,
                           const float* __restrict__ b2, float* __restrict__ gate, int N) {
  int wid = threadIdx.x >> 5, lane = threadIdx.x & 31;
  int node = blockIdx.x * 8 + wid;
  if (node >= N) return;
  float s = 0.f;
  #pragma unroll
  for (int k = 0; k < 4; ++k) {
    int c = lane + k * 32;
    s += u[(long long)node * HD + c] * w2[c];
  }
  #pragma unroll
  for (int off = 16; off > 0; off >>= 1) s += __shfl_xor(s, off, 32);
  if (lane == 0) gate[node] = 1.f / (1.f + expf(-(s + b2[0])));
}

__global__ void k_pool(const float* __restrict__ h, const float* __restrict__ gate,
                       const int* __restrict__ batch, float* __restrict__ g, int N) {
  long long i = (long long)blockIdx.x * blockDim.x + threadIdx.x;
  if (i >= (long long)N * HD) return;
  int n = (int)(i >> 7), c = (int)(i & (HD - 1));
  atomicAdd(&g[(long long)batch[n] * HD + c], h[i] * gate[n]);
}

// BN over pooled [256 x 128] + convert to f16 GEMM operand, single block
__global__ void k_bn_pool_f16(const float* __restrict__ g, const float* __restrict__ gamma,
                              const float* __restrict__ beta, _Float16* __restrict__ out,
                              int rows) {
  int c = threadIdx.x;
  float s = 0.f, s2 = 0.f;
  for (int r = 0; r < rows; ++r) { float v = g[r * HD + c]; s += v; s2 += v * v; }
  float mean = s / rows, var = s2 / rows - mean * mean;
  float sc = gamma[c] * rsqrtf(var + 1e-5f);
  float sh = beta[c] - mean * sc;
  for (int r = 0; r < rows; ++r) out[r * HD + c] = (_Float16)(g[r * HD + c] * sc + sh);
}

// classifier [256x128] @ [128x10] + bias, log_softmax
__global__ __launch_bounds__(256) void k_classify(const float* __restrict__ gfc,
    const float* __restrict__ W, const float* __restrict__ b,
    float* __restrict__ out, int G) {
  __shared__ float sW[HD * 10];
  __shared__ float sb[10];
  for (int i = threadIdx.x; i < HD * 10; i += blockDim.x) sW[i] = W[i];
  if (threadIdx.x < 10) sb[threadIdx.x] = b[threadIdx.x];
  __syncthreads();
  int gi = blockIdx.x * blockDim.x + threadIdx.x;
  if (gi >= G) return;
  float lg[10];
  #pragma unroll
  for (int j = 0; j < 10; ++j) lg[j] = sb[j];
  for (int c = 0; c < HD; ++c) {
    float v = gfc[(long long)gi * HD + c];
    #pragma unroll
    for (int j = 0; j < 10; ++j) lg[j] += v * sW[c * 10 + j];
  }
  float mx = lg[0];
  #pragma unroll
  for (int j = 1; j < 10; ++j) mx = fmaxf(mx, lg[j]);
  float se = 0.f;
  #pragma unroll
  for (int j = 0; j < 10; ++j) se += expf(lg[j] - mx);
  float lse = mx + logf(se);
  #pragma unroll
  for (int j = 0; j < 10; ++j) out[(long long)gi * 10 + j] = lg[j] - lse;
}

// ------------------------------------------------------------------- host
extern "C" void kernel_launch(void* const* d_in, const int* in_sizes, int n_in,
                              void* d_out, int out_size, void* d_ws, size_t ws_size,
                              hipStream_t stream) {
  (void)n_in; (void)out_size; (void)ws_size;
  const float* x           = (const float*)d_in[0];
  const int*   edges       = (const int*)d_in[1];
  const int*   batch       = (const int*)d_in[2];
  const float* bn_feat_g   = (const float*)d_in[3];
  const float* bn_feat_b   = (const float*)d_in[4];
  const float* conv_feat_W = (const float*)d_in[5];
  const float* conv_feat_b = (const float*)d_in[6];
  const float* conv_W      = (const float*)d_in[7];
  const float* conv_b      = (const float*)d_in[8];
  const float* bn_conv_g   = (const float*)d_in[9];
  const float* bn_conv_b   = (const float*)d_in[10];
  const float* gate_W1     = (const float*)d_in[11];
  const float* gate_b1     = (const float*)d_in[12];
  const float* gate_W2     = (const float*)d_in[13];
  const float* gate_b2     = (const float*)d_in[14];
  const float* fc_W        = (const float*)d_in[15];
  const float* fc_b        = (const float*)d_in[16];
  const float* bn_fc_g     = (const float*)d_in[17];
  const float* bn_fc_b     = (const float*)d_in[18];
  const float* cls_W       = (const float*)d_in[19];
  const float* cls_b       = (const float*)d_in[20];

  const int N    = in_sizes[0] / HD;
  const int E    = in_sizes[1] / 2;
  const int Npad = (N + 127) & ~127;
  const int G    = 256;

  char* ws = (char*)d_ws;
  size_t off = 0;
  auto carve = [&](size_t bytes) -> char* {
    char* p = ws + off;
    off += (bytes + 255) & ~(size_t)255;
    return p;
  };
  float*     h     = (float*)carve((size_t)Npad * HD * 4);
  float*     t     = (float*)carve((size_t)Npad * HD * 4);
  _Float16*  Ah    = (_Float16*)carve((size_t)Npad * HD * 2);
  _Float16*  Wt    = (_Float16*)carve((size_t)HD * HD * 2);
  float*     dinv  = (float*)carve((size_t)Npad * 4);
  float*     stats = (float*)carve((size_t)2 * HD * 4);
  float*     gate  = (float*)carve((size_t)Npad * 4);
  float*     gpool = (float*)carve((size_t)G * HD * 4);

  const int* srcIdx = edges;
  const int* dstIdx = edges + E;

  dim3 b256(256);
  auto cdiv = [](long long a, long long b) { return (int)((a + b - 1) / b); };
  const long long NE = (long long)Npad * HD;
  const int RPB = 512;

  // degrees + dinv
  k_zero<<<cdiv(Npad, 256), b256, 0, stream>>>(dinv, Npad);
  k_edge_deg<<<cdiv(E, 256), b256, 0, stream>>>(dstIdx, dinv, E);
  k_dinv<<<cdiv(N, 256), b256, 0, stream>>>(dinv, N);

  auto conv_layer = [&](const float* inp, const float* bng, const float* bnb,
                        const float* W, const float* bias) {
    k_zero<<<1, b256, 0, stream>>>(stats, 2 * HD);
    k_bn_stats<<<cdiv(N, RPB), dim3(HD), 0, stream>>>(inp, stats, N, RPB);
    k_bn_finalize<<<1, dim3(HD), 0, stream>>>(stats, bng, bnb, 1.0f / (float)N);
    k_bn_apply_f16<<<cdiv(NE, 256), b256, 0, stream>>>(inp, stats, Ah, N, Npad);
    k_wcvt<<<HD * HD / 256, b256, 0, stream>>>(W, Wt);
    k_gemm_wmma<<<Npad / 128, b256, 0, stream>>>(Ah, Wt, nullptr, t, 0);
    k_zero<<<cdiv(NE, 256), b256, 0, stream>>>(h, NE);
    k_scatter<<<cdiv((long long)E * 32, 256), b256, 0, stream>>>(srcIdx, dstIdx, dinv, t, h, E);
    k_finalize<<<cdiv((long long)N * HD, 256), b256, 0, stream>>>(h, t, dinv, bias, N);
  };

  conv_layer(x, bn_feat_g, bn_feat_b, conv_feat_W, conv_feat_b);
  for (int i = 0; i < 3; ++i)
    conv_layer(h, bn_conv_g + i * HD, bn_conv_b + i * HD,
               conv_W + (size_t)i * HD * HD, conv_b + i * HD);

  // gate = sigmoid(relu(h @ W1 + b1) @ W2 + b2)
  k_to_f16<<<cdiv(NE, 256), b256, 0, stream>>>(h, Ah, N, Npad);
  k_wcvt<<<HD * HD / 256, b256, 0, stream>>>(gate_W1, Wt);
  k_gemm_wmma<<<Npad / 128, b256, 0, stream>>>(Ah, Wt, gate_b1, t, 1);
  k_gate_dot<<<cdiv(N, 8), b256, 0, stream>>>(t, gate_W2, gate_b2, gate, N);

  // pooled g = segment_sum(h * gate, batch)
  k_zero<<<cdiv((long long)G * HD, 256), b256, 0, stream>>>(gpool, (long long)G * HD);
  k_pool<<<cdiv((long long)N * HD, 256), b256, 0, stream>>>(h, gate, batch, gpool, N);

  // g = relu(BN(g) @ fc_W + fc_b)
  k_bn_pool_f16<<<1, dim3(HD), 0, stream>>>(gpool, bn_fc_g, bn_fc_b, Ah, G);
  k_wcvt<<<HD * HD / 256, b256, 0, stream>>>(fc_W, Wt);
  k_gemm_wmma<<<G / 128, b256, 0, stream>>>(Ah, Wt, fc_b, t, 1);

  // logits + log_softmax
  k_classify<<<1, b256, 0, stream>>>(t, cls_W, cls_b, (float*)d_out, G);
}
